// WF_V2_84018150244900
// MI455X (gfx1250) — matrix operand order repaired and checked by reading, not compiled, not run on hardware
//
#include <hip/hip_runtime.h>
#include <stdint.h>

// ---------------------------------------------------------------------------
// channel-attention: out = alph * (atten @ X) + x,  atten = (X X^T)/(n n^T+eps)
// X = x[b] as (256, 4096) fp32.  bf16x3 split GEMMs on v_wmma_f32_16x16x32_bf16
// 2x2 register blocking per wave: 12 WMMA per 8 LDS fragments per 32-K step.
// ---------------------------------------------------------------------------

typedef __attribute__((ext_vector_type(16))) __bf16 v16bf;
typedef __attribute__((ext_vector_type(8)))  float  v8f;

#define CCH 256
#define HW  4096
#define NB  32
#define LDP 66        // LDS row stride in bf16 elems (33 dwords, odd -> no bank conflicts)
#define EPSV 1e-8f

union FragU { v16bf v; uint32_t u[8]; };

__device__ __forceinline__ uint32_t f2bf(float f) {           // fp32 -> bf16 bits (RNE)
  uint32_t u = __float_as_uint(f);
  return (u + 0x7FFFu + ((u >> 16) & 1u)) >> 16;
}
__device__ __forceinline__ float bf2f(uint32_t h) { return __uint_as_float(h << 16); }

__device__ __forceinline__ void split2(float f0, float f1, uint32_t& hpack, uint32_t& lpack) {
  uint32_t h0 = f2bf(f0), h1 = f2bf(f1);
  uint32_t l0 = f2bf(f0 - bf2f(h0)), l1 = f2bf(f1 - bf2f(h1));
  hpack = h0 | (h1 << 16);
  lpack = l0 | (l1 << 16);
}

// A operand 16x32 bf16: half=0 -> K 0..7 (v0..3), 16..23 (v4..7); half=1 -> +8
__device__ __forceinline__ v16bf fragA(const uint16_t* row, int half) {
  const uint32_t* p = (const uint32_t*)row;
  FragU f;
#pragma unroll
  for (int v = 0; v < 4; ++v) f.u[v] = p[4 * half + v];
#pragma unroll
  for (int v = 0; v < 4; ++v) f.u[4 + v] = p[8 + 4 * half + v];
  return f.v;
}
// B operand 32x16 bf16: half=0 -> K 0..7 (v0..3), 16..23 (v4..7); half=1 -> +8
__device__ __forceinline__ v16bf fragB(const uint16_t* row, int half) {
  const uint32_t* p = (const uint32_t*)row;
  FragU f;
#pragma unroll
  for (int v = 0; v < 4; ++v) f.u[v] = p[4 * half + v];
#pragma unroll
  for (int v = 0; v < 4; ++v) f.u[4 + v] = p[8 + 4 * half + v];
  return f.v;
}

// bf16x3: acc += aL*bH + aH*bL + aH*bH   (f32 accumulate)
__device__ __forceinline__ v8f mm3(v16bf aH, v16bf aL, v16bf bH, v16bf bL, v8f acc) {
  acc = __builtin_amdgcn_wmma_f32_16x16x32_bf16(false, aL, false, bH, (short)0, acc, false, false);
  acc = __builtin_amdgcn_wmma_f32_16x16x32_bf16(false, aH, false, bL, (short)0, acc, false, false);
  acc = __builtin_amdgcn_wmma_f32_16x16x32_bf16(false, aH, false, bH, (short)0, acc, false, false);
  return acc;
}

// ------------------------- K1: G = X X^T (per batch) ------------------------
// Workgroup tile: 128 (i) x 64 (j).  8 waves in 4x2 grid, each 32x32 (2x2 tiles).
__global__ __launch_bounds__(256) void gram_k(const float* __restrict__ x,
                                              float* __restrict__ G) {
  __shared__ uint16_t Ahi[128 * LDP], Alo[128 * LDP], Bhi[64 * LDP], Blo[64 * LDP];
  const int b  = blockIdx.z;
  const int i0 = blockIdx.y * 128;
  const int j0 = blockIdx.x * 64;
  const float* X = x + (size_t)b * CCH * HW;

  const int tid  = threadIdx.x;
  const int lane = tid & 31;
  const int wave = tid >> 5;
  const int half = lane >> 4;
  const int l15  = lane & 15;
  const int wm   = wave >> 1;      // 0..3 : 32-row band
  const int wn   = wave & 1;       // 0..1 : 32-col band

  v8f acc00 = {}, acc01 = {}, acc10 = {}, acc11 = {};

  for (int kb = 0; kb < HW; kb += 64) {
    __syncthreads();
#pragma unroll
    for (int p = 0; p < 16; ++p) {           // A: 128 rows x 64 k
      int lin = p * 256 + tid;
      int r = lin >> 5;
      int c = (lin & 31) * 2;
      const float2 va = *(const float2*)(X + (size_t)(i0 + r) * HW + kb + c);
      uint32_t h, l;
      split2(va.x, va.y, h, l);
      *(uint32_t*)&Ahi[r * LDP + c] = h;
      *(uint32_t*)&Alo[r * LDP + c] = l;
      if (kb + 64 < HW)
        __builtin_prefetch(X + (size_t)(i0 + r) * HW + kb + 64 + c, 0, 1);
    }
#pragma unroll
    for (int p = 0; p < 8; ++p) {            // B: 64 rows x 64 k
      int lin = p * 256 + tid;
      int r = lin >> 5;
      int c = (lin & 31) * 2;
      const float2 vb = *(const float2*)(X + (size_t)(j0 + r) * HW + kb + c);
      uint32_t h, l;
      split2(vb.x, vb.y, h, l);
      *(uint32_t*)&Bhi[r * LDP + c] = h;
      *(uint32_t*)&Blo[r * LDP + c] = l;
      if (kb + 64 < HW)
        __builtin_prefetch(X + (size_t)(j0 + r) * HW + kb + 64 + c, 0, 1);
    }
    __syncthreads();
#pragma unroll
    for (int kk = 0; kk < 2; ++kk) {
      const int a0 = (wm * 32 + l15) * LDP + kk * 32;
      const int a1 = (wm * 32 + 16 + l15) * LDP + kk * 32;
      const int b0 = (wn * 32 + l15) * LDP + kk * 32;
      const int b1 = (wn * 32 + 16 + l15) * LDP + kk * 32;
      v16bf aH0 = fragA(&Ahi[a0], half), aL0 = fragA(&Alo[a0], half);
      v16bf aH1 = fragA(&Ahi[a1], half), aL1 = fragA(&Alo[a1], half);
      v16bf bH0 = fragB(&Bhi[b0], half), bL0 = fragB(&Blo[b0], half);
      v16bf bH1 = fragB(&Bhi[b1], half), bL1 = fragB(&Blo[b1], half);
      acc00 = mm3(aH0, aL0, bH0, bL0, acc00);
      acc01 = mm3(aH0, aL0, bH1, bL1, acc01);
      acc10 = mm3(aH1, aL1, bH0, bL0, acc10);
      acc11 = mm3(aH1, aL1, bH1, bL1, acc11);
    }
  }

  float* Gb = G + (size_t)b * CCH * CCH;
#pragma unroll
  for (int v = 0; v < 8; ++v) {
    int m0 = i0 + wm * 32 + v + 8 * half;
    int m1 = m0 + 16;
    int c0 = j0 + wn * 32 + l15;
    int c1 = c0 + 16;
    Gb[(size_t)m0 * CCH + c0] = acc00[v];
    Gb[(size_t)m0 * CCH + c1] = acc01[v];
    Gb[(size_t)m1 * CCH + c0] = acc10[v];
    Gb[(size_t)m1 * CCH + c1] = acc11[v];
  }
}

// ------------------------- K2: per-channel norms ---------------------------
__global__ void norm_k(const float* __restrict__ G, float* __restrict__ nrm) {
  int i = blockIdx.x * blockDim.x + threadIdx.x;
  if (i < NB * CCH) {
    int b = i >> 8, c = i & 255;
    nrm[i] = sqrtf(G[(size_t)b * CCH * CCH + (size_t)c * (CCH + 1)]);
  }
}

// ------------ K3: out = alph * (atten @ X) + x  (normalization fused) ------
// Workgroup tile: 128 (out channel) x 64 (spatial). Same 2x2 wave blocking.
__global__ __launch_bounds__(256) void attn_k(const float* __restrict__ x,
                                              const float* __restrict__ G,
                                              const float* __restrict__ nrm,
                                              const float* __restrict__ alph,
                                              float* __restrict__ out) {
  __shared__ uint16_t Ahi[128 * LDP], Alo[128 * LDP], Bhi[64 * LDP], Blo[64 * LDP];
  const int b  = blockIdx.z;
  const int i0 = blockIdx.y * 128;    // output channel tile
  const int n0 = blockIdx.x * 64;     // spatial tile
  const float* X  = x + (size_t)b * CCH * HW;
  const float* Gb = G + (size_t)b * CCH * CCH;
  const float* nb = nrm + b * CCH;

  const int tid  = threadIdx.x;
  const int lane = tid & 31;
  const int wave = tid >> 5;
  const int half = lane >> 4;
  const int l15  = lane & 15;
  const int wm   = wave >> 1;
  const int wn   = wave & 1;

  v8f acc00 = {}, acc01 = {}, acc10 = {}, acc11 = {};

  for (int kb = 0; kb < CCH; kb += 64) {
    __syncthreads();
    // atten tile: 128 rows m x 64 cols k(=channel j), normalize on load
#pragma unroll
    for (int p = 0; p < 16; ++p) {
      int lin = p * 256 + tid;
      int r = lin >> 5;
      int c = (lin & 31) * 2;
      float g0 = Gb[(size_t)(i0 + r) * CCH + kb + c];
      float g1 = Gb[(size_t)(i0 + r) * CCH + kb + c + 1];
      float ni = nb[i0 + r];
      float a0 = g0 / (ni * nb[kb + c]     + EPSV);
      float a1 = g1 / (ni * nb[kb + c + 1] + EPSV);
      uint32_t h, l;
      split2(a0, a1, h, l);
      *(uint32_t*)&Ahi[r * LDP + c] = h;
      *(uint32_t*)&Alo[r * LDP + c] = l;
    }
    // X tile transposed into LDS as [n][k] so B-operand k-pairs are dword-contiguous
#pragma unroll
    for (int p = 0; p < 16; ++p) {
      int lin = p * 256 + tid;
      int kr = lin >> 6;
      int cn = lin & 63;
      float f = X[(size_t)(kb + kr) * HW + n0 + cn];
      uint32_t h = f2bf(f);
      uint32_t l = f2bf(f - bf2f(h));
      Bhi[cn * LDP + kr] = (uint16_t)h;
      Blo[cn * LDP + kr] = (uint16_t)l;
      if (kb + 64 < CCH)
        __builtin_prefetch(X + (size_t)(kb + 64 + kr) * HW + n0 + cn, 0, 1);
    }
    __syncthreads();
#pragma unroll
    for (int kk = 0; kk < 2; ++kk) {
      const int a0 = (wm * 32 + l15) * LDP + kk * 32;
      const int a1 = (wm * 32 + 16 + l15) * LDP + kk * 32;
      const int b0 = (wn * 32 + l15) * LDP + kk * 32;
      const int b1 = (wn * 32 + 16 + l15) * LDP + kk * 32;
      v16bf aH0 = fragA(&Ahi[a0], half), aL0 = fragA(&Alo[a0], half);
      v16bf aH1 = fragA(&Ahi[a1], half), aL1 = fragA(&Alo[a1], half);
      v16bf bH0 = fragB(&Bhi[b0], half), bL0 = fragB(&Blo[b0], half);
      v16bf bH1 = fragB(&Bhi[b1], half), bL1 = fragB(&Blo[b1], half);
      acc00 = mm3(aH0, aL0, bH0, bL0, acc00);
      acc01 = mm3(aH0, aL0, bH1, bL1, acc01);
      acc10 = mm3(aH1, aL1, bH0, bL0, acc10);
      acc11 = mm3(aH1, aL1, bH1, bL1, acc11);
    }
  }

  const float av = alph[0];
#pragma unroll
  for (int v = 0; v < 8; ++v) {
    int m0 = i0 + wm * 32 + v + 8 * half;
    int m1 = m0 + 16;
    int c0 = n0 + wn * 32 + l15;
    int c1 = c0 + 16;
    size_t base = (size_t)b * CCH * HW;
    size_t r0 = base + (size_t)m0 * HW;
    size_t r1 = base + (size_t)m1 * HW;
    out[r0 + c0] = av * acc00[v] + x[r0 + c0];
    out[r0 + c1] = av * acc01[v] + x[r0 + c1];
    out[r1 + c0] = av * acc10[v] + x[r1 + c0];
    out[r1 + c1] = av * acc11[v] + x[r1 + c1];
  }
}

// ---------------------------------------------------------------------------
extern "C" void kernel_launch(void* const* d_in, const int* in_sizes, int n_in,
                              void* d_out, int out_size, void* d_ws, size_t ws_size,
                              hipStream_t stream) {
  const float* x    = (const float*)d_in[0];
  const float* alph = (const float*)d_in[1];
  float* out = (float*)d_out;

  float* G   = (float*)d_ws;                         // 32*256*256 f32 = 8 MB
  float* nrm = G + (size_t)NB * CCH * CCH;           // + 8192 f32

  dim3 blk(256);
  gram_k<<<dim3(CCH / 64, CCH / 128, NB), blk, 0, stream>>>(x, G);
  norm_k<<<dim3((NB * CCH) / 256), blk, 0, stream>>>(G, nrm);
  attn_k<<<dim3(HW / 64, CCH / 128, NB), blk, 0, stream>>>(x, G, nrm, alph, out);
}
